// SparseResBlock_6880537608517
// MI455X (gfx1250) — matrix operand order repaired and checked by reading, not compiled, not run on hardware
//
#include <hip/hip_runtime.h>
#include <hip/hip_bf16.h>
#include <stdint.h>

// ---------------- problem constants ----------------
#define BATCH      4
#define GRID       64
#define NPB        65536          // points per batch (batch = point >> 16)
#define NPTS       (BATCH * NPB)  // 262144
#define CCH        64             // channels
#define GROUPS     32
#define CG         2              // channels per group
#define KOFF       27
#define EMB        512
#define GN_CNT     ((float)(NPB * CG))   // 131072 elements per (batch,group)

typedef __attribute__((ext_vector_type(16))) _Float16 v16h;
typedef __attribute__((ext_vector_type(8)))  _Float16 v8h;
typedef __attribute__((ext_vector_type(2)))  _Float16 v2h;
typedef __attribute__((ext_vector_type(8)))  float    v8f;

union AFrag { v16h v; v8h p[2]; };

// ---------------- workspace layout (bytes) ----------------
#define WS_LOOKUP   0u                          // B*G^3 int32      = 4,194,304
#define WS_STATS1   4194304u                    // 256 f32 (+pad)
#define WS_STATS2   4195328u                    // 256 f32 (+pad)
#define WS_EMBOUT   4196352u                    // 256 f32 (+pad)
#define WS_W1SW     4197376u                    // 27*4096 f16 = 221,184
#define WS_W2SW     4418560u                    // 221,184
#define WS_HBUF     4639744u                    // N*C f16 = 33,554,432
#define WS_MID      38194176u                   // N*C f32 = 67,108,864
// total ~105.3 MB

// ---------------- kernel 0: init lookup (-1) + zero stats ----------------
__global__ void k_init(int* __restrict__ lookup, float* __restrict__ stats /*512 f32*/) {
    int i = blockIdx.x * blockDim.x + threadIdx.x;
    if (i < BATCH * GRID * GRID * GRID) lookup[i] = -1;
    if (i < 512) stats[i] = 0.0f;
}

// ---------------- kernel 1: scatter point indices into lookup ----------------
__global__ void k_scatter(const int* __restrict__ coords, int* __restrict__ lookup) {
    int i = blockIdx.x * blockDim.x + threadIdx.x;
    if (i >= NPTS) return;
    int b = coords[i * 4 + 0];
    int x = coords[i * 4 + 1];
    int y = coords[i * 4 + 2];
    int z = coords[i * 4 + 3];
    lookup[(((b * GRID + x) * GRID + y) * GRID + z)] = i;
}

// ---------------- kernel 2: group-norm stats (sum, sumsq per batch,group) ----------------
// block = 256 threads = (group g = t%32, row pr = t/32); block covers 128 consecutive points
__global__ void k_gn_stats(const float* __restrict__ x, float* __restrict__ stats) {
    __shared__ float ls[8][GROUPS][2];
    const int t  = threadIdx.x;
    const int g  = t & 31;
    const int pr = t >> 5;
    const int base = blockIdx.x * 128;           // 128 | NPB, so single batch per block
    float s = 0.f, ss = 0.f;
    for (int p = pr; p < 128; p += 8) {
        const float* row = x + (size_t)(base + p) * CCH + g * CG;
        float a = row[0], b = row[1];
        s  += a + b;
        ss += a * a + b * b;
    }
    ls[pr][g][0] = s; ls[pr][g][1] = ss;
    __syncthreads();
    if (pr == 0) {
        for (int r = 1; r < 8; ++r) { s += ls[r][g][0]; ss += ls[r][g][1]; }
        const int batch = base >> 16;            // NPB == 65536
        atomicAdd(&stats[(batch * GROUPS + g) * 2 + 0], s);
        atomicAdd(&stats[(batch * GROUPS + g) * 2 + 1], ss);
    }
}

// ---------------- kernel 3: apply GN + SiLU, emit f16 rows ----------------
__global__ void k_gn_apply_silu(const float* __restrict__ x, const float* __restrict__ stats,
                                const float* __restrict__ gw, const float* __restrict__ gb,
                                _Float16* __restrict__ out) {
    int tid = blockIdx.x * blockDim.x + threadIdx.x;   // one per (point, group)
    if (tid >= NPTS * GROUPS) return;
    const int point = tid >> 5;
    const int g     = tid & 31;
    const int batch = point >> 16;
    const float s   = stats[(batch * GROUPS + g) * 2 + 0];
    const float ss  = stats[(batch * GROUPS + g) * 2 + 1];
    const float mean = s / GN_CNT;
    const float var  = ss / GN_CNT - mean * mean;
    const float inv  = rsqrtf(var + 1e-5f);
    const float2 xv = *(const float2*)(x + (size_t)point * CCH + g * CG);
    float v0 = (xv.x - mean) * inv * gw[g * 2 + 0] + gb[g * 2 + 0];
    float v1 = (xv.y - mean) * inv * gw[g * 2 + 1] + gb[g * 2 + 1];
    v0 = v0 / (1.0f + __expf(-v0));
    v1 = v1 / (1.0f + __expf(-v1));
    v2h o; o.x = (_Float16)v0; o.y = (_Float16)v1;
    *(v2h*)(out + (size_t)point * CCH + g * CG) = o;
}

// ---------------- kernel 4: embedding GEMV: silu(emb) @ emb_w.T + emb_b ----------------
__global__ void k_emb(const float* __restrict__ emb, const float* __restrict__ emb_w,
                      const float* __restrict__ emb_b, float* __restrict__ embout) {
    int t = threadIdx.x;                 // 256 threads: b = t/64, c = t%64
    int b = t >> 6, c = t & 63;
    float acc = 0.f;
    for (int e = 0; e < EMB; ++e) {
        float ev = emb[b * EMB + e];
        float sv = ev / (1.0f + __expf(-ev));
        acc += sv * emb_w[c * EMB + e];
    }
    embout[b * CCH + c] = acc + emb_b[c];
}

// ---------------- kernel 5: weight convert + swizzle into B-fragment-major f16 ----------------
// out[i]: i = k*4096 + (c*4+t)*512 + lane*16 + e
//   cin  = c*32 + (e/8)*16 + (lane/16)*8 + (e%8)   (B-matrix 16-bit layout: lanes 0-15 K=0..7,16..23; lanes 16-31 K=8..15,24..31)
//   cout = t*16 + (lane%16)
__global__ void k_swizzle_w(const float* __restrict__ W, _Float16* __restrict__ out) {
    int i = blockIdx.x * blockDim.x + threadIdx.x;
    if (i >= KOFF * 4096) return;
    const int e    = i & 15;
    const int lane = (i >> 4) & 31;
    const int f    = (i >> 9) & 7;       // c*4+t
    const int t    = f & 3;
    const int c    = f >> 2;
    const int k    = i >> 12;
    const int cin  = c * 32 + (e >> 3) * 16 + (lane >> 4) * 8 + (e & 7);
    const int cout = t * 16 + (lane & 15);
    out[i] = (_Float16)W[(size_t)k * (CCH * CCH) + cin * CCH + cout];
}

// ---------------- kernel 6: sparse 3^3 conv via WMMA ----------------
// one wave32 per 32-point tile (two 16-point M-subtiles); each B fragment feeds
// two WMMAs so weight-load issue is amortized 2x. Full 32x64 f32 output lives
// in 8x v8f accumulators (64 VGPRs).
template<bool ADD_EMB, bool ADD_RESID>
__global__ void k_conv_wmma(const _Float16* __restrict__ hbuf, const int* __restrict__ lookup,
                            const int* __restrict__ coords, const _Float16* __restrict__ wsw,
                            const float* __restrict__ bias, const float* __restrict__ embrow,
                            const float* __restrict__ resid, float* __restrict__ out) {
    const int gtid  = blockIdx.x * blockDim.x + threadIdx.x;
    const int lane  = gtid & 31;
    const int tileBase = (gtid >> 5) * 32;          // 32 points per wave
    const int nn = lane & 15;                        // output column within 16-tile
    const int mh = (lane >> 4) << 3;                 // C/D row half offset (0 or 8)
    const int h8 = (lane >> 4) << 3;                 // A-fragment K sub-offset (0 or 8)

    // this lane's A rows: point tileBase + (lane&15) and tileBase + 16 + (lane&15)
    const int4 cc0 = *(const int4*)(coords + (size_t)(tileBase + nn) * 4);
    const int4 cc1 = *(const int4*)(coords + (size_t)(tileBase + 16 + nn) * 4);
    const int pb = cc0.x;                            // whole tile is one batch (32 | NPB)

    // accumulators initialized with conv bias (same value down each column)
    v8f acc[2][4];
#pragma unroll
    for (int t = 0; t < 4; ++t) {
        const float bv = bias[t * 16 + nn];
        v8f a;
#pragma unroll
        for (int i = 0; i < 8; ++i) a[i] = bv;
        acc[0][t] = a;
        acc[1][t] = a;
    }

    const v16h vzero = {};

#pragma unroll 1
    for (int k = 0; k < KOFF; ++k) {
        const int dx = k / 9 - 1, dy = (k / 3) % 3 - 1, dz = k % 3 - 1;

        // neighbor indices for both M-subtiles
        int idx0 = -1, idx1 = -1;
        {
            const unsigned nx = (unsigned)(cc0.y + dx);
            const unsigned ny = (unsigned)(cc0.z + dy);
            const unsigned nz = (unsigned)(cc0.w + dz);
            if (nx < GRID && ny < GRID && nz < GRID)
                idx0 = lookup[(((pb << 6) | (int)nx) << 6 | (int)ny) << 6 | (int)nz];
        }
        {
            const unsigned nx = (unsigned)(cc1.y + dx);
            const unsigned ny = (unsigned)(cc1.z + dy);
            const unsigned nz = (unsigned)(cc1.w + dz);
            if (nx < GRID && ny < GRID && nz < GRID)
                idx1 = lookup[(((pb << 6) | (int)nx) << 6 | (int)ny) << 6 | (int)nz];
        }

        // gather A fragments: a[m][c] = 16x32 f16 chunk for subtile m, K-chunk c
        AFrag a00, a01, a10, a11;
        {
            const _Float16* arow = hbuf + (size_t)(idx0 < 0 ? 0 : idx0) * CCH;
            a00.p[0] = *(const v8h*)(arow + h8);
            a00.p[1] = *(const v8h*)(arow + h8 + 16);
            a01.p[0] = *(const v8h*)(arow + 32 + h8);
            a01.p[1] = *(const v8h*)(arow + 32 + h8 + 16);
            if (idx0 < 0) { a00.v = vzero; a01.v = vzero; }
        }
        {
            const _Float16* arow = hbuf + (size_t)(idx1 < 0 ? 0 : idx1) * CCH;
            a10.p[0] = *(const v8h*)(arow + h8);
            a10.p[1] = *(const v8h*)(arow + h8 + 16);
            a11.p[0] = *(const v8h*)(arow + 32 + h8);
            a11.p[1] = *(const v8h*)(arow + 32 + h8 + 16);
            if (idx1 < 0) { a10.v = vzero; a11.v = vzero; }
        }

        const _Float16* wk = wsw + (size_t)k * 4096 + (size_t)lane * 16;
#pragma unroll
        for (int c = 0; c < 2; ++c) {
#pragma unroll
            for (int t = 0; t < 4; ++t) {
                AFrag bu;
                const _Float16* bp = wk + (c * 4 + t) * 512;
                bu.p[0] = *(const v8h*)(bp);
                bu.p[1] = *(const v8h*)(bp + 8);
                acc[0][t] = __builtin_amdgcn_wmma_f32_16x16x32_f16(
                    false, (c ? a01.v : a00.v), false, bu.v,
                    (short)0, acc[0][t], false, false);
                acc[1][t] = __builtin_amdgcn_wmma_f32_16x16x32_f16(
                    false, (c ? a11.v : a10.v), false, bu.v,
                    (short)0, acc[1][t], false, false);
            }
        }
    }

    // store 32x64 f32 tile (+ optional per-batch emb row, + optional residual)
    const int batch = tileBase >> 16;
#pragma unroll
    for (int m = 0; m < 2; ++m) {
        const int rowBase = tileBase + m * 16 + mh;
#pragma unroll
        for (int t = 0; t < 4; ++t) {
            const int col = t * 16 + nn;
            float addc = 0.f;
            if (ADD_EMB) addc = embrow[batch * CCH + col];
#pragma unroll
            for (int v = 0; v < 8; ++v) {
                const int row = rowBase + v;
                float val = acc[m][t][v] + addc;
                if (ADD_RESID) val += resid[(size_t)row * CCH + col];
                out[(size_t)row * CCH + col] = val;
            }
        }
    }
}

// ---------------- launch ----------------
extern "C" void kernel_launch(void* const* d_in, const int* in_sizes, int n_in,
                              void* d_out, int out_size, void* d_ws, size_t ws_size,
                              hipStream_t stream) {
    const float* feats   = (const float*)d_in[0];
    const int*   coords  = (const int*)  d_in[1];
    const float* emb     = (const float*)d_in[2];
    const float* gn1_w   = (const float*)d_in[3];
    const float* gn1_b   = (const float*)d_in[4];
    const float* conv1_w = (const float*)d_in[5];
    const float* conv1_b = (const float*)d_in[6];
    const float* emb_w   = (const float*)d_in[7];
    const float* emb_b   = (const float*)d_in[8];
    const float* gn2_w   = (const float*)d_in[9];
    const float* gn2_b   = (const float*)d_in[10];
    const float* conv2_w = (const float*)d_in[11];
    const float* conv2_b = (const float*)d_in[12];

    uint8_t* ws = (uint8_t*)d_ws;
    int*       lookup = (int*)      (ws + WS_LOOKUP);
    float*     stats1 = (float*)    (ws + WS_STATS1);
    float*     stats2 = (float*)    (ws + WS_STATS2);
    float*     embout = (float*)    (ws + WS_EMBOUT);
    _Float16*  w1sw   = (_Float16*) (ws + WS_W1SW);
    _Float16*  w2sw   = (_Float16*) (ws + WS_W2SW);
    _Float16*  hbuf   = (_Float16*) (ws + WS_HBUF);
    float*     mid    = (float*)    (ws + WS_MID);
    float*     outp   = (float*)    d_out;

    const int LUT = BATCH * GRID * GRID * GRID;

    // 0) lookup = -1, stats = 0
    k_init<<<(LUT + 255) / 256, 256, 0, stream>>>(lookup, stats1 /* stats1+stats2 contiguous 512 */);
    // 1) scatter
    k_scatter<<<(NPTS + 255) / 256, 256, 0, stream>>>(coords, lookup);
    // 2) weight swizzle + embedding GEMV (independent)
    k_swizzle_w<<<(KOFF * 4096 + 255) / 256, 256, 0, stream>>>(conv1_w, w1sw);
    k_swizzle_w<<<(KOFF * 4096 + 255) / 256, 256, 0, stream>>>(conv2_w, w2sw);
    k_emb<<<1, 256, 0, stream>>>(emb, emb_w, emb_b, embout);
    // 3) GN1 stats -> apply+SiLU -> f16
    k_gn_stats<<<NPTS / 128, 256, 0, stream>>>(feats, stats1);
    k_gn_apply_silu<<<(NPTS * GROUPS) / 256, 256, 0, stream>>>(feats, stats1, gn1_w, gn1_b, hbuf);
    // 4) conv1 (WMMA) + emb row -> mid (f32)
    {
        const int waves = NPTS / 32;                 // 8192 waves, 32 points each
        k_conv_wmma<true, false><<<waves * 32 / 256, 256, 0, stream>>>(
            hbuf, lookup, coords, w1sw, conv1_b, embout, nullptr, mid);
    }
    // 5) GN2 stats -> apply+SiLU -> f16 (reuse hbuf)
    k_gn_stats<<<NPTS / 128, 256, 0, stream>>>(mid, stats2);
    k_gn_apply_silu<<<(NPTS * GROUPS) / 256, 256, 0, stream>>>(mid, stats2, gn2_w, gn2_b, hbuf);
    // 6) conv2 (WMMA) + residual -> d_out (f32)
    {
        const int waves = NPTS / 32;
        k_conv_wmma<false, true><<<waves * 32 / 256, 256, 0, stream>>>(
            hbuf, lookup, coords, w2sw, conv2_b, nullptr, feats, outp);
    }
    (void)in_sizes; (void)n_in; (void)out_size; (void)ws_size;
}
